// DenseGRUODE_32933809225772
// MI455X (gfx1250) — matrix operand order repaired and checked by reading, compiled, not run on hardware
//
#include <hip/hip_runtime.h>

typedef __attribute__((ext_vector_type(16))) _Float16 v16h;
typedef __attribute__((ext_vector_type(8)))  float    v8f;

#define T_STEPS 1000
#define BATCH   256
#define DIN     64
#define DOUT    128
#define DT_EULER 0.05f

#define WMMA_F16(a, b, c) \
    __builtin_amdgcn_wmma_f32_16x16x32_f16(false, (a), false, (b), (short)0, (c), false, false)

// sigmoid via v_exp_f32 + v_rcp_f32 (avoid the IEEE div_scale/div_fixup sequence)
__device__ __forceinline__ float sigmoid_fast(float x) {
    return __builtin_amdgcn_rcpf(1.0f + __expf(-x));
}

// tanh(y) = 1 - 2 / (exp(2y) + 1), again with raw v_rcp_f32
__device__ __forceinline__ float tanh_fast(float y) {
    return 1.0f - 2.0f * __builtin_amdgcn_rcpf(__expf(2.0f * y) + 1.0f);
}

// Load a 16x32 f16 A-fragment from row-major LDS [16][128].
// ISA layout: lane<16 -> M=lane, K={klo..klo+7, klo+16..klo+23}
//             lane>=16 -> same M=lane-16, K offset +8
__device__ __forceinline__ v16h load_afrag(const _Float16* p) {
    v16h a;
    *(float4*)&a         = *(const float4*)(p);        // halves 0..7  (8 f16 = 16B)
    *(((float4*)&a) + 1) = *(const float4*)(p + 16);   // halves 8..15
    return a;
}

__global__ __launch_bounds__(256)
void gruode_kernel(const float* __restrict__ X,
                   const float* __restrict__ Whr, const float* __restrict__ bhr,
                   const float* __restrict__ Whz, const float* __restrict__ bhz,
                   const float* __restrict__ Whh, const float* __restrict__ bhh,
                   const float* __restrict__ h0,
                   float* __restrict__ out)
{
    // h (f16) and r*h (f16) staged in LDS in A-matrix-friendly row-major layout
    __shared__ __align__(16) _Float16 hA[16 * DOUT];
    __shared__ __align__(16) _Float16 rhA[16 * DOUT];

    const int tid   = threadIdx.x;
    const int lane  = tid & 31;
    const int wid   = tid >> 5;           // 0..7 : N-tile owned by this wave
    const int lrow  = lane & 15;          // tile-local batch row (A/C layout)
    const int lhi   = lane >> 4;          // half-wave selector
    const int ncol  = wid * 16 + lrow;    // output column 0..127 (B/C layout: N = lane%16)
    const int bBase = blockIdx.x * 16;    // batch row base for this workgroup
    const int kbB   = lhi * 16;           // unused; B K base per lane half is 8*lhi (koff)

    // ---------------- one-time: weight B-fragments -> registers (persist all T steps)
    // B layout (32x16 f16): lanes 0-15 hold K={0..7,16..23} for column N=lane, lanes 16-31 hold K={8..15,24..31}
    v16h Br[6], Bz[6], Bh[6];
#pragma unroll
    for (int c = 0; c < 6; ++c) {
        const int koff = c * 32 + lhi * 8; // W row index (0..191); rows 0..127 = h part, 128..191 = x part
#pragma unroll
        for (int j = 0; j < 16; ++j) {
            Br[c][j] = (_Float16)Whr[(koff + j + (j & 8)) * DOUT + ncol];
            Bz[c][j] = (_Float16)Whz[(koff + j + (j & 8)) * DOUT + ncol];
            Bh[c][j] = (_Float16)Whh[(koff + j + (j & 8)) * DOUT + ncol];
        }
    }
    const float br = bhr[ncol];
    const float bz = bhz[ncol];
    const float bh = bhh[ncol];

    // ---------------- init h : broadcast h0 over batch rows
    const float h0v = h0[ncol];
    v8f hc;                                // f32 h in C layout: comp i -> row (lhi? i+8 : i), col ncol
#pragma unroll
    for (int i = 0; i < 8; ++i) hc[i] = h0v;
#pragma unroll
    for (int i = 0; i < 8; ++i) {
        const int row = lhi ? (i + 8) : i;
        hA[row * DOUT + ncol] = (_Float16)h0v;
    }

    for (int t = 0; t < T_STEPS; ++t) {
        __syncthreads();                   // hA for this step is complete

        // ---- x_t A-fragments (K = 128..191 of hx), straight from global (L2-resident)
        v16h xf[2];
#pragma unroll
        for (int c = 0; c < 2; ++c) {
            const float* xp = X + (((size_t)t * BATCH) + bBase + lrow) * DIN + c * 32 + lhi * 8;
            float4 p0 = *(const float4*)(xp);
            float4 p1 = *(const float4*)(xp + 4);
            float4 p2 = *(const float4*)(xp + 16);
            float4 p3 = *(const float4*)(xp + 20);
            float tmp[16];
            *(float4*)&tmp[0]  = p0;
            *(float4*)&tmp[4]  = p1;
            *(float4*)&tmp[8]  = p2;
            *(float4*)&tmp[12] = p3;
#pragma unroll
            for (int j = 0; j < 16; ++j) xf[c][j] = (_Float16)tmp[j];
        }

        // ---- r and z gates: acc = bias; 4 h-chunks + 2 x-chunks of K=32
        v8f ar, az;
#pragma unroll
        for (int i = 0; i < 8; ++i) { ar[i] = br; az[i] = bz; }
#pragma unroll
        for (int c = 0; c < 4; ++c) {
            v16h a = load_afrag(&hA[lrow * DOUT + c * 32 + lhi * 8]);
            ar = WMMA_F16(a, Br[c], ar);
            az = WMMA_F16(a, Bz[c], az);
        }
#pragma unroll
        for (int c = 0; c < 2; ++c) {
            ar = WMMA_F16(xf[c], Br[4 + c], ar);
            az = WMMA_F16(xf[c], Bz[4 + c], az);
        }

        // ---- r = sigmoid(ar); stage r*h (f16) for the u GEMM
#pragma unroll
        for (int i = 0; i < 8; ++i) {
            const float r   = sigmoid_fast(ar[i]);
            const int   row = lhi ? (i + 8) : i;
            rhA[row * DOUT + ncol] = (_Float16)(r * hc[i]);
        }
        __syncthreads();                   // rhA complete

        // ---- u = tanh([r*h, x] @ Whh + bh)
        v8f au;
#pragma unroll
        for (int i = 0; i < 8; ++i) au[i] = bh;
#pragma unroll
        for (int c = 0; c < 4; ++c) {
            v16h a = load_afrag(&rhA[lrow * DOUT + c * 32 + lhi * 8]);
            au = WMMA_F16(a, Bh[c], au);
        }
#pragma unroll
        for (int c = 0; c < 2; ++c) {
            au = WMMA_F16(xf[c], Bh[4 + c], au);
        }

        // ---- Euler update in f32, stream h_new to output, refresh f16 hA for next step
#pragma unroll
        for (int i = 0; i < 8; ++i) {
            const float z  = sigmoid_fast(az[i]);
            const float u  = tanh_fast(au[i]);
            const float hn = hc[i] + (1.0f - z) * (u - hc[i]) * DT_EULER;
            hc[i] = hn;
            const int row = lhi ? (i + 8) : i;
            out[(((size_t)(bBase + row)) * T_STEPS + t) * DOUT + ncol] = hn;
            hA[row * DOUT + ncol] = (_Float16)hn;
        }
    }
}

extern "C" void kernel_launch(void* const* d_in, const int* in_sizes, int n_in,
                              void* d_out, int out_size, void* d_ws, size_t ws_size,
                              hipStream_t stream) {
    const float* X   = (const float*)d_in[0];
    const float* Whr = (const float*)d_in[1];
    const float* bhr = (const float*)d_in[2];
    const float* Whz = (const float*)d_in[3];
    const float* bhz = (const float*)d_in[4];
    const float* Whh = (const float*)d_in[5];
    const float* bhh = (const float*)d_in[6];
    const float* h0  = (const float*)d_in[7];
    float* out = (float*)d_out;

    gruode_kernel<<<BATCH / 16, 256, 0, stream>>>(X, Whr, bhr, Whz, bhz, Whh, bhh, h0, out);
}
